// GRU2D_naive_71760313582275
// MI455X (gfx1250) — matrix operand order repaired and checked by reading, not compiled, not run on hardware
//
#include <hip/hip_runtime.h>

typedef _Float16 half_t;
typedef __attribute__((ext_vector_type(16))) half_t v16h;
typedef __attribute__((ext_vector_type(8)))  half_t v8h;
typedef __attribute__((ext_vector_type(8)))  float  v8f;

#define NROWS 262144
#define DIM   128   // HID_DIM == IN_DIM == 128
#define KDIM  256   // concat width

// ---------------------------------------------------------------------------
// One-time weight repack: W[g] (256x128 f32, row-major K x N) -> f16 WMMA
// B-fragments, lane-contiguous:  wpk[((g*8+nt)*8+kt)*512 + lane*16 + i]
//   n = nt*16 + (lane&15),  k = kt*32 + (i>>3)*16 + (lane>>4)*8 + (i&7)
// g: 0 = W_r, 1 = W_z, 2 = W_n.  Total 3*8*8*512 = 98304 halves (192 KB).
// ---------------------------------------------------------------------------
__global__ void pack_weights_k(const float* __restrict__ Wr,
                               const float* __restrict__ Wz,
                               const float* __restrict__ Wn,
                               half_t* __restrict__ wpk) {
  int t      = blockIdx.x * 256 + threadIdx.x;   // 0..98303
  int fragid = t >> 9;
  int within = t & 511;
  int lane   = within >> 4;
  int i      = within & 15;
  int g      = fragid >> 6;
  int rem    = fragid & 63;
  int nt     = rem >> 3;
  int kt     = rem & 7;
  int n      = nt * 16 + (lane & 15);
  int k      = kt * 32 + (i >> 3) * 16 + (lane >> 4) * 8 + (i & 7);
  const float* W = (g == 0) ? Wr : (g == 1) ? Wz : Wn;
  wpk[t] = (half_t)W[k * DIM + n];
}

// Load one pre-packed B fragment: 32 contiguous bytes per lane (coalesced 1KB).
__device__ __forceinline__ v16h load_bfrag(const half_t* __restrict__ wpk,
                                           int g, int nt, int kt, int lane) {
  const half_t* p = wpk + (((g * 8 + nt) * 8 + kt) << 9) + lane * 16;
  v8h lo = *(const v8h*)p;
  v8h hi = *(const v8h*)(p + 8);
  v16h v;
#pragma unroll
  for (int j = 0; j < 8; ++j) { v[j] = lo[j]; v[8 + j] = hi[j]; }
  return v;
}

__device__ __forceinline__ float sigmoid_f(float x) {
  return 1.0f / (1.0f + __expf(-x));
}

// ---------------------------------------------------------------------------
// Fused GRU cell. Block = 256 threads = 8 waves; wave handles 16 rows.
// ---------------------------------------------------------------------------
__launch_bounds__(256)
__global__ void gru_wmma_k(const float* __restrict__ inF,
                           const float* __restrict__ hidF,
                           const float* __restrict__ br,
                           const float* __restrict__ bz,
                           const float* __restrict__ bn,
                           const half_t* __restrict__ wpk,
                           float* __restrict__ out) {
  __shared__ alignas(16) half_t upd[8][16 * DIM];   // per-wave r*h tile (f16), 32 KB

  const int lane = threadIdx.x & 31;
  const int w    = threadIdx.x >> 5;
  const int mlo  = lane & 15;     // A row / B,C col within tile
  const int hh   = lane >> 4;     // lane-half select
  const int rowbase = blockIdx.x * 128 + w * 16;

  // ---- Load A fragments for X = [hidden | input], f32 -> f16 ----
  // 16-bit A layout: lane (hh,mlo): VGPR0..3 -> K = 8*hh + 0..7,
  //                                 VGPR4..7 -> K = 16 + 8*hh + 0..7  (per 32-K tile)
  v16h a[8];
#pragma unroll
  for (int kt = 0; kt < 8; ++kt) {
    const float* src = (kt < 4)
        ? (hidF + (size_t)(rowbase + mlo) * DIM + kt * 32)
        : (inF  + (size_t)(rowbase + mlo) * DIM + (kt - 4) * 32);
    float4 c0a = *(const float4*)(src + 8 * hh);
    float4 c0b = *(const float4*)(src + 8 * hh + 4);
    float4 c1a = *(const float4*)(src + 16 + 8 * hh);
    float4 c1b = *(const float4*)(src + 16 + 8 * hh + 4);
    v16h av;
    av[0]  = (half_t)c0a.x; av[1]  = (half_t)c0a.y;
    av[2]  = (half_t)c0a.z; av[3]  = (half_t)c0a.w;
    av[4]  = (half_t)c0b.x; av[5]  = (half_t)c0b.y;
    av[6]  = (half_t)c0b.z; av[7]  = (half_t)c0b.w;
    av[8]  = (half_t)c1a.x; av[9]  = (half_t)c1a.y;
    av[10] = (half_t)c1a.z; av[11] = (half_t)c1a.w;
    av[12] = (half_t)c1b.x; av[13] = (half_t)c1b.y;
    av[14] = (half_t)c1b.z; av[15] = (half_t)c1b.w;
    a[kt] = av;
  }

  // ---- Phase 1: r and z gates, write r*h (f16) to LDS, keep z in VGPRs ----
  float zsave[8][8];
#pragma unroll
  for (int nt = 0; nt < 8; ++nt) {
    v8f accr = {};
    v8f accz = {};
#pragma unroll
    for (int kt = 0; kt < 8; ++kt) {
      v16h bf_r = load_bfrag(wpk, 0, nt, kt, lane);
      v16h bf_z = load_bfrag(wpk, 1, nt, kt, lane);
      accr = __builtin_amdgcn_wmma_f32_16x16x32_f16(false, a[kt], false, bf_r,
                                                    (short)0, accr, false, false);
      accz = __builtin_amdgcn_wmma_f32_16x16x32_f16(false, a[kt], false, bf_z,
                                                    (short)0, accz, false, false);
    }
    const int col = nt * 16 + mlo;
    const float brs = br[col];
    const float bzs = bz[col];
#pragma unroll
    for (int i = 0; i < 8; ++i) {
      const int m  = i + 8 * hh;                 // C/D layout: VGPR i -> row i+8*hh
      float rv = sigmoid_f(accr[i] + brs);
      float zv = sigmoid_f(accz[i] + bzs);
      zsave[nt][i] = zv;
      float hv = hidF[(size_t)(rowbase + m) * DIM + col];
      upd[w][m * DIM + col] = (half_t)(rv * hv);
    }
  }

  __syncthreads();   // LDS writes are cross-lane consumed below

  // ---- Phase 2: n gate from [r*h | input], then blend + store ----
#pragma unroll
  for (int nt = 0; nt < 8; ++nt) {
    v8f accn = {};
#pragma unroll
    for (int kt = 0; kt < 8; ++kt) {
      v16h af;
      if (kt < 4) {
        // Rebuild A fragment from LDS (16B-aligned ds_load_b128 chunks)
        const half_t* p = &upd[w][mlo * DIM + kt * 32 + 8 * hh];
        v8h lo = *(const v8h*)p;         // K = kt*32 + 8*hh + 0..7
        v8h hi = *(const v8h*)(p + 16);  // K = kt*32 + 16 + 8*hh + 0..7
#pragma unroll
        for (int j = 0; j < 8; ++j) { af[j] = lo[j]; af[8 + j] = hi[j]; }
      } else {
        af = a[kt];                      // input half reused directly
      }
      v16h bf_n = load_bfrag(wpk, 2, nt, kt, lane);
      accn = __builtin_amdgcn_wmma_f32_16x16x32_f16(false, af, false, bf_n,
                                                    (short)0, accn, false, false);
    }
    const int col = nt * 16 + mlo;
    const float bns = bn[col];
#pragma unroll
    for (int i = 0; i < 8; ++i) {
      const int m  = i + 8 * hh;
      float q  = tanhf(accn[i] + bns);
      float hv = hidF[(size_t)(rowbase + m) * DIM + col];
      float zv = zsave[nt][i];
      out[(size_t)(rowbase + m) * DIM + col] = (1.0f - zv) * hv + zv * q;
    }
  }
}

// ---------------------------------------------------------------------------
extern "C" void kernel_launch(void* const* d_in, const int* in_sizes, int n_in,
                              void* d_out, int out_size, void* d_ws, size_t ws_size,
                              hipStream_t stream) {
  const float* inF  = (const float*)d_in[0];   // input_feat   [N,128]
  const float* hidF = (const float*)d_in[1];   // hidden_feat  [N,128]
  // d_in[2..5] (weight embeddings / dirs) are unused by the reference
  const float* Wz = (const float*)d_in[6];
  const float* bz = (const float*)d_in[7];
  const float* Wr = (const float*)d_in[8];
  const float* br = (const float*)d_in[9];
  const float* Wn = (const float*)d_in[10];
  const float* bn = (const float*)d_in[11];
  half_t* wpk = (half_t*)d_ws;                 // 192 KB packed f16 weights

  pack_weights_k<<<384, 256, 0, stream>>>(Wr, Wz, Wn, wpk);
  gru_wmma_k<<<NROWS / 128, 256, 0, stream>>>(inF, hidF, br, bz, bn, wpk,
                                              (float*)d_out);
}
